// DNCS2Q_81217831567537
// MI455X (gfx1250) — matrix operand order repaired and checked by reading, not compiled, not run on hardware
//
#include <hip/hip_runtime.h>
#include <hip/hip_bf16.h>

// ---------------------------------------------------------------------------
// DNC forward for MI455X (gfx1250, wave32, WMMA).
// Design:
//   * dnc_prep : convert weights fp32->f16 in workspace, init memory/link.
//   * dnc_scan : persistent kernel, 16 blocks (1 per batch), 512 thr (16 waves).
//                All recurrent state in LDS; controller matvecs + link einsums
//                + read vectors via v_wmma_f32_16x16x32_f16; allocation via
//                bitonic sort + scan; memory/link streamed through L2.
//   * dnc_out  : output GEMM out[b,v,m] = mem[b,m,:]·Wfc[v,:] + bfc[v],
//                one 16x16 tile per wave, 4 chained WMMAs over K=128.
// ---------------------------------------------------------------------------

typedef __attribute__((ext_vector_type(16))) _Float16 v16h;
typedef __attribute__((ext_vector_type(8)))  float    v8f;

#define DELTA  1e-6f
#define CLIPV  20.0f
#define RH     4
#define MSLOT  128
#define BATCH  16
#define TSTEP  128
#define DDIM   128
#define WDIM   128
#define VOCAB  4096
#define IN0    (DDIM + RH*WDIM)              // 640
#define G4D    (4*DDIM)                      // 512
#define ISZ    (RH*WDIM + 3*WDIM + 5*RH + 3) // 919
#define ISZP   928                           // padded to 58*16

__device__ __forceinline__ float sigf(float x)  { return 1.f/(1.f+__expf(-x)); }
__device__ __forceinline__ float splusf(float x){ return fmaxf(x,0.f)+log1pf(__expf(-fabsf(x))); }

// y[16] += W[o0:o0+16, :K] @ vec[:K]  via chained WMMA f32<-f16.
// A = vec chunk broadcast into all 16 rows (all D rows identical);
// B = 16 weight columns, each lane loads 16 contiguous f16 (32B aligned).
// After return: outputs j=0..15 live in acc[0] of lanes (lane&15)==j.
__device__ __forceinline__ v8f wmma_matvec(const _Float16* __restrict__ Wblk, int ldk,
                                           const _Float16* __restrict__ vec, int K,
                                           v8f acc, int lane) {
  const int g   = lane >> 4;   // 0: K 0-15 (A and B); 1: other half
  const int col = lane & 15;
  const _Float16* wrow = Wblk + (size_t)col*ldk + g*16;
  for (int kc = 0; kc < K; kc += 32) {
    v16h a;
#pragma unroll
    for (int q = 0; q < 8; ++q) {
      a[q]   = vec[kc + g*16 + q];
      a[8+q] = vec[kc + g*16 + 8 + q];
    }
    v16h bm = *(const v16h*)(wrow + kc);
    __builtin_prefetch(wrow + kc + 512, 0, 1);   // global_prefetch_b8, L2-warm next rows
    acc = __builtin_amdgcn_wmma_f32_16x16x32_f16(false, a, false, bm, (short)0, acc, false, false);
  }
  return acc;
}

// ---------------------------------------------------------------------------
__global__ void dnc_prep(const float* __restrict__ Wih0, const float* __restrict__ Whh0,
                         const float* __restrict__ Wih1, const float* __restrict__ Whh1,
                         const float* __restrict__ Wint, const float* __restrict__ Wfc,
                         _Float16* h0, _Float16* h1, _Float16* h2, _Float16* h3,
                         _Float16* h4, _Float16* h5,
                         float* mem, float* link) {
  const long long N0 = (long long)G4D*IN0;       // 327680
  const long long N1 = (long long)G4D*DDIM;      // 65536
  const long long N4 = (long long)ISZP*DDIM;     // 118784
  const long long N5 = (long long)VOCAB*WDIM;    // 524288
  const long long N6 = (long long)BATCH*MSLOT*WDIM; // 262144
  const long long total = N0 + 3*N1 + N4 + N5 + 2*N6;
  for (long long i = (long long)blockIdx.x*blockDim.x + threadIdx.x; i < total;
       i += (long long)gridDim.x*blockDim.x) {
    long long j = i;
    if (j < N0) { h0[j] = (_Float16)Wih0[j]; continue; } j -= N0;
    if (j < N1) { h1[j] = (_Float16)Whh0[j]; continue; } j -= N1;
    if (j < N1) { h2[j] = (_Float16)Wih1[j]; continue; } j -= N1;
    if (j < N1) { h3[j] = (_Float16)Whh1[j]; continue; } j -= N1;
    if (j < N4) { int o = (int)(j/DDIM), k = (int)(j%DDIM);
                  h4[j] = (_Float16)((o < ISZ) ? Wint[(size_t)o*DDIM + k] : 0.f); continue; } j -= N4;
    if (j < N5) { h5[j] = (_Float16)Wfc[j]; continue; } j -= N5;
    if (j < N6) { mem[j] = DELTA; continue; } j -= N6;
    link[j] = 0.f;
  }
}

// ---------------------------------------------------------------------------
__global__ __launch_bounds__(512) void dnc_scan(
    const int*   __restrict__ input, const float* __restrict__ emb,
    const _Float16* __restrict__ Wih0h, const _Float16* __restrict__ Whh0h,
    const _Float16* __restrict__ Wih1h, const _Float16* __restrict__ Whh1h,
    const _Float16* __restrict__ Winth,
    const float* __restrict__ bih0, const float* __restrict__ bhh0,
    const float* __restrict__ bih1, const float* __restrict__ bhh1,
    const float* __restrict__ bint,
    float* __restrict__ memG, float* __restrict__ linkG)
{
  __shared__ _Float16 s_Lh[MSLOT*MSLOT];     // 32KB f16 mirror of link[b,0,:,:]
  __shared__ _Float16 s_inp_h[IN0];          // [x_t | last_read] f16
  __shared__ _Float16 s_h0h[DDIM], s_h1h[DDIM], s_outh[DDIM];
  __shared__ _Float16 s_rwh[RH*MSLOT];
  __shared__ float s_g[G4D];
  __shared__ float s_xi[ISZP];
  __shared__ float s_c0[DDIM], s_c1[DDIM];
  __shared__ float s_rk[RH*WDIM], s_rkb[RH], s_wkey[WDIM], s_erase[WDIM], s_wvec[WDIM];
  __shared__ float s_freeg[RH], s_modes[RH*3];
  __shared__ float s_usage[MSLOT], s_ww[MSLOT], s_prec[MSLOT];
  __shared__ float s_rw[RH*MSLOT], s_cw[RH*MSLOT], s_fwd[RH*MSLOT], s_bwd[RH*MSLOT];
  __shared__ float s_alloc[MSLOT], s_wcw[MSLOT];
  __shared__ float s_sv[MSLOT]; __shared__ int s_si[MSLOT];
  __shared__ float s_scan[MSLOT];
  __shared__ float s_n2a[MSLOT], s_n2b[MSLOT];
  __shared__ float s_scal[8];   // 0=wbeta 1=alloc_g 2=write_g 3=sum_ww 4..7=|rk_r|+d

  const int tid  = threadIdx.x;
  const int lane = tid & 31;
  const int wid  = tid >> 5;
  const int b    = blockIdx.x;
  float* mem = memG  + (size_t)b*MSLOT*WDIM;
  float* lnk = linkG + (size_t)b*MSLOT*MSLOT;

  // ---- initial carry ----
  for (int i = tid; i < IN0; i += 512) s_inp_h[i] = (_Float16)0.f;
  if (tid < DDIM) { s_h0h[tid] = (_Float16)0.f; s_h1h[tid] = (_Float16)0.f;
                    s_c0[tid] = 0.f; s_c1[tid] = 0.f; }
  if (tid < MSLOT) { s_usage[tid] = 0.f; s_ww[tid] = DELTA; s_prec[tid] = 0.f;
                     s_n2a[tid] = (float)WDIM * DELTA * DELTA; }
  for (int i = tid; i < RH*MSLOT; i += 512) s_rw[i] = DELTA;
  __syncthreads();

  for (int t = 0; t < TSTEP; ++t) {
    // 0: embedding lookup into x part of inp
    if (tid < DDIM) {
      int tok = input[b*TSTEP + t];
      s_inp_h[tid] = (_Float16)emb[(size_t)tok*DDIM + tid];
    }
    __syncthreads();

    // 1: layer0 gates  g = Wih0@inp + Whh0@h0 + b   (WMMA, 32 out-groups / 16 waves)
    for (int og = wid; og < G4D/16; og += 16) {
      int o0 = og*16;
      v8f acc = {};
      acc = wmma_matvec(Wih0h + (size_t)o0*IN0,  IN0,  s_inp_h, IN0,  acc, lane);
      acc = wmma_matvec(Whh0h + (size_t)o0*DDIM, DDIM, s_h0h,   DDIM, acc, lane);
      if (lane < 16) { int o = o0 + lane; s_g[o] = acc[0] + bih0[o] + bhh0[o]; }
    }
    __syncthreads();

    // 2: LSTM cell 0
    if (tid < DDIM) {
      float gi = s_g[tid], gf = s_g[DDIM+tid], gg = s_g[2*DDIM+tid], go = s_g[3*DDIM+tid];
      float c = sigf(gf)*s_c0[tid] + sigf(gi)*tanhf(gg);
      float h = sigf(go)*tanhf(c);
      s_c0[tid] = c; s_h0h[tid] = (_Float16)h;
    }
    __syncthreads();

    // 3: layer1 gates
    for (int og = wid; og < G4D/16; og += 16) {
      int o0 = og*16;
      v8f acc = {};
      acc = wmma_matvec(Wih1h + (size_t)o0*DDIM, DDIM, s_h0h, DDIM, acc, lane);
      acc = wmma_matvec(Whh1h + (size_t)o0*DDIM, DDIM, s_h1h, DDIM, acc, lane);
      if (lane < 16) { int o = o0 + lane; s_g[o] = acc[0] + bih1[o] + bhh1[o]; }
    }
    __syncthreads();

    // 4: LSTM cell 1 (+ clip for interface input)
    if (tid < DDIM) {
      float gi = s_g[tid], gf = s_g[DDIM+tid], gg = s_g[2*DDIM+tid], go = s_g[3*DDIM+tid];
      float c = sigf(gf)*s_c1[tid] + sigf(gi)*tanhf(gg);
      float h = sigf(go)*tanhf(c);
      s_c1[tid] = c; s_h1h[tid] = (_Float16)h;
      s_outh[tid] = (_Float16)fminf(fmaxf(h, -CLIPV), CLIPV);
    }
    __syncthreads();

    // 5: interface projection xi = Wint@out + bint  (58 groups, padded)
    for (int og = wid; og < ISZP/16; og += 16) {
      int o0 = og*16;
      v8f acc = {};
      acc = wmma_matvec(Winth + (size_t)o0*DDIM, DDIM, s_outh, DDIM, acc, lane);
      if (lane < 16) { int o = o0 + lane;
        s_xi[o] = acc[0] + ((o < ISZ) ? bint[o] : 0.f); }
    }
    __syncthreads();

    // 6: interface nonlinearities + f16 copy of OLD rw + zero new norms
    for (int i = tid; i < ISZ; i += 512) {
      float v = s_xi[i];
      if      (i < 512)  s_rk[i]        = tanhf(v);
      else if (i < 516)  s_rkb[i-512]   = splusf(v);
      else if (i < 644)  s_wkey[i-516]  = tanhf(v);
      else if (i == 644) s_scal[0]      = splusf(v);
      else if (i < 773)  s_erase[i-645] = sigf(v);
      else if (i < 901)  s_wvec[i-773]  = tanhf(v);
      else if (i < 905)  s_freeg[i-901] = sigf(v);
      else if (i == 905) s_scal[1]      = sigf(v);
      else if (i == 906) s_scal[2]      = sigf(v);
    }
    if (tid < RH) {  // read modes: softmax over 3
      float a0 = s_xi[907+3*tid], a1 = s_xi[908+3*tid], a2 = s_xi[909+3*tid];
      float mx = fmaxf(a0, fmaxf(a1, a2));
      float e0 = __expf(a0-mx), e1 = __expf(a1-mx), e2 = __expf(a2-mx);
      float s = e0+e1+e2;
      s_modes[tid*3] = e0/s; s_modes[tid*3+1] = e1/s; s_modes[tid*3+2] = e2/s;
    }
    for (int i = tid; i < RH*MSLOT; i += 512) s_rwh[i] = (_Float16)s_rw[i];
    if (tid < MSLOT) s_n2b[tid] = 0.f;
    __syncthreads();

    // 7: usage update (uses prev ww/rw) + write-key dots vs OLD memory + sort keys
    if (tid < MSLOT) {
      float u = s_usage[tid];
      u = u + (1.f - u) * s_ww[tid];                 // single write head
      float f = 1.f;
#pragma unroll
      for (int r = 0; r < RH; ++r) f *= (1.f - s_freeg[r]*s_rw[r*MSLOT+tid]);
      u *= f;
      s_usage[tid] = u;
      s_sv[tid] = DELTA + (1.f - DELTA)*u;
      s_si[tid] = tid;
      const float* mrow = mem + (size_t)tid*WDIM;
      float d = 0.f;
      for (int w = 0; w < WDIM; ++w) d += s_wkey[w]*mrow[w];
      s_wcw[tid] = d;
    }
    __syncthreads();

    // 8: write content softmax (serial, M=128)
    if (tid == 0) {
      float kn = 0.f;
      for (int w = 0; w < WDIM; ++w) kn += s_wkey[w]*s_wkey[w];
      kn = sqrtf(kn) + DELTA;
      float beta = s_scal[0], mx = -1e30f;
      for (int m = 0; m < MSLOT; ++m) {
        float sc = s_wcw[m] / ((sqrtf(s_n2a[m]) + DELTA) * kn) * beta;
        s_wcw[m] = sc; mx = fmaxf(mx, sc);
      }
      float sum = 0.f;
      for (int m = 0; m < MSLOT; ++m) { float e = __expf(s_wcw[m]-mx); s_wcw[m] = e; sum += e; }
      float inv = 1.f/sum;
      for (int m = 0; m < MSLOT; ++m) s_wcw[m] *= inv;
    }
    __syncthreads();

    // 9: bitonic sort (stable argsort semantics via (val,idx) key)
    for (int k = 2; k <= MSLOT; k <<= 1)
      for (int j = k >> 1; j > 0; j >>= 1) {
        if (tid < MSLOT) {
          int ixj = tid ^ j;
          if (ixj > tid) {
            bool up = ((tid & k) == 0);
            float v0 = s_sv[tid], v1 = s_sv[ixj];
            int   i0 = s_si[tid], i1 = s_si[ixj];
            bool gt = (v0 > v1) || (v0 == v1 && i0 > i1);
            if (gt == up) { s_sv[tid] = v1; s_sv[ixj] = v0; s_si[tid] = i1; s_si[ixj] = i0; }
          }
        }
        __syncthreads();
      }

    // 10: exclusive cumprod over sorted u, scatter allocation weights
    if (tid < MSLOT) s_scan[tid] = s_sv[tid];
    __syncthreads();
    for (int off = 1; off < MSLOT; off <<= 1) {
      float v = 0.f;
      if (tid < MSLOT) { v = s_scan[tid]; if (tid >= off) v *= s_scan[tid-off]; }
      __syncthreads();
      if (tid < MSLOT) s_scan[tid] = v;
      __syncthreads();
    }
    if (tid < MSLOT) {
      float excl = (tid == 0) ? 1.f : s_scan[tid-1];
      s_alloc[s_si[tid]] = (1.f - s_sv[tid]) * excl;
    }
    __syncthreads();

    // 11: write weights
    if (tid < MSLOT) {
      float ag = s_scal[1], wg = s_scal[2];
      s_ww[tid] = wg * (ag*s_alloc[tid] + (1.f-ag)*s_wcw[tid]);
    }
    __syncthreads();
    // 12: sum(ww)
    if (tid == 0) { float s = 0.f; for (int m = 0; m < MSLOT; ++m) s += s_ww[m]; s_scal[3] = s; }
    __syncthreads();

    // 13: memory update (stream through L2) + fused new row norms (ds_add_f32)
    {
      int m  = tid >> 2;
      int w0 = (tid & 3) * 32;
      float wwm = s_ww[m], part = 0.f;
      float* mrow = mem + (size_t)m*WDIM;
      for (int q = 0; q < 32; ++q) {
        int w = w0 + q;
        float v = mrow[w];
        v = v*(1.f - wwm*s_erase[w]) + wwm*s_wvec[w];
        mrow[w] = v;
        part += v*v;
      }
      atomicAdd(&s_n2b[m], part);
    }
    __syncthreads();

    // 14: link update (stream) + f16 mirror to LDS for the WMMA einsums
    for (int e = tid; e < MSLOT*MSLOT; e += 512) {
      int i = e >> 7, j = e & 127;
      float nv = (1.f - s_ww[i] - s_ww[j]) * lnk[e] + s_ww[i]*s_prec[j];
      if (i == j) nv = 0.f;
      lnk[e] = nv;
      s_Lh[e] = (_Float16)nv;
    }
    __syncthreads();

    // 15: precedence update, publish new norms, read-key norms
    if (tid < MSLOT) {
      s_prec[tid] = (1.f - s_scal[3]) * s_prec[tid] + s_ww[tid];
      s_n2a[tid]  = s_n2b[tid];
    }
    if (tid >= 128 && tid < 128 + RH) {
      int r = tid - 128; float kn = 0.f;
      for (int w = 0; w < WDIM; ++w) { float v = s_rk[r*WDIM+w]; kn += v*v; }
      s_scal[4+r] = sqrtf(kn) + DELTA;
    }
    __syncthreads();

    // 16: read content logits vs NEW memory (one (r,m) pair per thread)
    {
      int r = tid & 3, m = tid >> 2;
      const float* mrow = mem + (size_t)m*WDIM;
      const float* key  = s_rk + r*WDIM;
      float d = 0.f;
      for (int w = 0; w < WDIM; ++w) d += key[w]*mrow[w];
      s_cw[r*MSLOT+m] = d / ((sqrtf(s_n2a[m]) + DELTA) * s_scal[4+r]) * s_rkb[r];
    }
    __syncthreads();
    // 17: per-head softmax
    if (tid < RH) {
      float* cw = s_cw + tid*MSLOT;
      float mx = -1e30f;
      for (int m = 0; m < MSLOT; ++m) mx = fmaxf(mx, cw[m]);
      float s = 0.f;
      for (int m = 0; m < MSLOT; ++m) { float e = __expf(cw[m]-mx); cw[m] = e; s += e; }
      float inv = 1.f/s;
      for (int m = 0; m < MSLOT; ++m) cw[m] *= inv;
    }
    __syncthreads();

    // 18: fwd = L @ rw_old^T (waves 0-7), bwd = rw_old @ L (waves 8-15) via WMMA
    {
      int g = lane >> 4, li = lane & 15;
      if (wid < 8) {
        int mt = wid;
        v8f acc = {};
        for (int kc = 0; kc < MSLOT; kc += 32) {
          v16h a, bm;
          int row = mt*16 + li;
#pragma unroll
          for (int q = 0; q < 8; ++q) {
            a[q]   = s_Lh[row*MSLOT + kc + g*16 + q];
            a[8+q] = s_Lh[row*MSLOT + kc + g*16 + 8 + q];
          }
#pragma unroll
          for (int q = 0; q < 16; ++q)
            bm[q] = (li < RH) ? s_rwh[li*MSLOT + kc + g*16 + q] : (_Float16)0.f;
          acc = __builtin_amdgcn_wmma_f32_16x16x32_f16(false, a, false, bm, (short)0, acc, false, false);
        }
        if (li < RH) {
          int ib = mt*16 + (g ? 8 : 0);
#pragma unroll
          for (int p = 0; p < 8; ++p) s_fwd[li*MSLOT + ib + p] = acc[p];
        }
      } else {
        int nt = wid - 8;
        v8f acc = {};
        for (int kc = 0; kc < MSLOT; kc += 32) {
          v16h a, bm;
#pragma unroll
          for (int q = 0; q < 8; ++q) {
            a[q]   = (li < RH) ? s_rwh[li*MSLOT + kc + g*16 + q]      : (_Float16)0.f;
            a[8+q] = (li < RH) ? s_rwh[li*MSLOT + kc + g*16 + 8 + q] : (_Float16)0.f;
          }
#pragma unroll
          for (int q = 0; q < 16; ++q)
            bm[q] = s_Lh[(kc + g*16 + q)*MSLOT + nt*16 + li];
          acc = __builtin_amdgcn_wmma_f32_16x16x32_f16(false, a, false, bm, (short)0, acc, false, false);
        }
        if (lane < 16) {
#pragma unroll
          for (int p = 0; p < RH; ++p) s_bwd[p*MSLOT + nt*16 + li] = acc[p];
        }
      }
    }
    __syncthreads();

    // 19: rw update
    for (int i = tid; i < RH*MSLOT; i += 512) {
      int r = i >> 7;
      float v = s_modes[r*3]*s_bwd[i] + s_modes[r*3+1]*s_fwd[i] + s_modes[r*3+2]*s_cw[i];
      s_rw[i] = v; s_rwh[i] = (_Float16)v;
    }
    __syncthreads();

    // 20: read vectors rv = rw_new @ memory via WMMA -> last_read (f16)
    if (wid < 8) {
      int w0 = wid*16;
      int g = lane >> 4, li = lane & 15;
      v8f acc = {};
      for (int kc = 0; kc < MSLOT; kc += 32) {
        v16h a, bm;
#pragma unroll
        for (int q = 0; q < 8; ++q) {
          a[q]   = (li < RH) ? s_rwh[li*MSLOT + kc + g*16 + q]      : (_Float16)0.f;
          a[8+q] = (li < RH) ? s_rwh[li*MSLOT + kc + g*16 + 8 + q] : (_Float16)0.f;
        }
#pragma unroll
        for (int q = 0; q < 16; ++q)
          bm[q] = (_Float16)mem[(size_t)(kc + g*16 + q)*WDIM + w0 + li];
        acc = __builtin_amdgcn_wmma_f32_16x16x32_f16(false, a, false, bm, (short)0, acc, false, false);
      }
      if (lane < 16) {
#pragma unroll
        for (int p = 0; p < RH; ++p)
          s_inp_h[DDIM + p*WDIM + w0 + li] = (_Float16)acc[p];
      }
    }
    __syncthreads();
  }
}

// ---------------------------------------------------------------------------
__global__ __launch_bounds__(256) void dnc_out(const _Float16* __restrict__ Wfch,
                                               const float* __restrict__ memG,
                                               const float* __restrict__ bfc,
                                               float* __restrict__ out) {
  const int wid  = threadIdx.x >> 5;
  const int lane = threadIdx.x & 31;
  const int id = blockIdx.x*8 + wid;     // 32768 tiles: b(16) x vt(256) x mt(8)
  const int mt = id & 7;
  const int vt = (id >> 3) & 255;
  const int b  = id >> 11;
  const int g = lane >> 4, li = lane & 15;
  const float* mb = memG + (size_t)b*MSLOT*WDIM;
  const _Float16* wr = Wfch + (size_t)(vt*16 + li)*WDIM;
  const float* mr = mb + (size_t)(mt*16 + li)*WDIM;
  v8f acc = {};
  for (int kc = 0; kc < WDIM; kc += 32) {
    v16h a, bm;
#pragma unroll
    for (int q = 0; q < 8; ++q) { a[q] = wr[kc + g*16 + q]; a[8+q] = wr[kc + g*16 + 8 + q]; }
#pragma unroll
    for (int q = 0; q < 16; ++q) bm[q] = (_Float16)mr[kc + g*16 + q];
    acc = __builtin_amdgcn_wmma_f32_16x16x32_f16(false, a, false, bm, (short)0, acc, false, false);
  }
  const int m = mt*16 + li;
  const int vbase = vt*16 + (g ? 8 : 0);
#pragma unroll
  for (int p = 0; p < 8; ++p) {
    int v = vbase + p;
    out[((size_t)b*VOCAB + v)*MSLOT + m] = acc[p] + bfc[v];
  }
}

// ---------------------------------------------------------------------------
extern "C" void kernel_launch(void* const* d_in, const int* in_sizes, int n_in,
                              void* d_out, int out_size, void* d_ws, size_t ws_size,
                              hipStream_t stream) {
  const int*   input = (const int*)  d_in[0];
  const float* emb   = (const float*)d_in[1];
  const float* Wih0  = (const float*)d_in[2];
  const float* Whh0  = (const float*)d_in[3];
  const float* bih0  = (const float*)d_in[4];
  const float* bhh0  = (const float*)d_in[5];
  const float* Wih1  = (const float*)d_in[6];
  const float* Whh1  = (const float*)d_in[7];
  const float* bih1  = (const float*)d_in[8];
  const float* bhh1  = (const float*)d_in[9];
  const float* Wint  = (const float*)d_in[10];
  const float* bint  = (const float*)d_in[11];
  const float* Wfc   = (const float*)d_in[12];
  const float* bfc   = (const float*)d_in[13];
  float* out = (float*)d_out;

  char* ws = (char*)d_ws;
  _Float16* hWih0 = (_Float16*)(ws + 0);        //  512*640*2 = 655360
  _Float16* hWhh0 = (_Float16*)(ws + 655360);   //  512*128*2 = 131072
  _Float16* hWih1 = (_Float16*)(ws + 786432);
  _Float16* hWhh1 = (_Float16*)(ws + 917504);
  _Float16* hWint = (_Float16*)(ws + 1048576);  //  928*128*2 = 237568
  _Float16* hWfc  = (_Float16*)(ws + 1286144);  // 4096*128*2 = 1048576
  float*    memS  = (float*)   (ws + 2334720);  // 16*128*128*4 = 1048576
  float*    linkS = (float*)   (ws + 3383296);  // 16*128*128*4 = 1048576
  // total workspace: ~4.43 MB

  dnc_prep<<<2048, 256, 0, stream>>>(Wih0, Whh0, Wih1, Whh1, Wint, Wfc,
                                     hWih0, hWhh0, hWih1, hWhh1, hWint, hWfc,
                                     memS, linkS);
  dnc_scan<<<BATCH, 512, 0, stream>>>(input, emb, hWih0, hWhh0, hWih1, hWhh1, hWint,
                                      bih0, bhh0, bih1, bhh1, bint, memS, linkS);
  dnc_out<<<4096, 256, 0, stream>>>(hWfc, memS, bfc, out);
}
